// MultiHeadAttentionBlock_87866440942209
// MI455X (gfx1250) — matrix operand order repaired and checked by reading, not compiled, not run on hardware
//
#include <hip/hip_runtime.h>

// ---------------------------------------------------------------------------
// MI455X / gfx1250 fused multi-head attention block (bf16 WMMA, f32 accum)
//   B=2, S=2048, E=1024, H=16, HD=64
//   pipeline: Q/K/V projections -> flash attention (causal) -> output proj
//   Softmax runs in exp2 domain: 1/sqrt(HD)*log2(e) is folded into the Q
//   projection; row-sum (l) is accumulated by a WMMA against an all-ones
//   B-matrix so it is rescaled identically to the O accumulator.
// ---------------------------------------------------------------------------

typedef __attribute__((ext_vector_type(16))) __bf16 v16bf;
typedef __attribute__((ext_vector_type(8)))  __bf16 v8bf;
typedef __attribute__((ext_vector_type(4)))  __bf16 v4bf;
typedef __attribute__((ext_vector_type(8)))  float  v8f;

#define B_  2
#define S_  2048
#define E_  1024
#define H_  16
#define HD_ 64

#define NEG_INF (-1.0e30f)

__device__ __forceinline__ __bf16 f2bf(float f) { return (__bf16)f; }  // RTNE cvt

__device__ __forceinline__ float fast_exp2(float x) {
#if __has_builtin(__builtin_amdgcn_exp2f)
  return __builtin_amdgcn_exp2f(x);     // v_exp_f32
#else
  return exp2f(x);
#endif
}

__device__ __forceinline__ v8f wmma_bf16(v16bf a, v16bf b, v8f c) {
  // v_wmma_f32_16x16x32_bf16  (D = A[16x32] * B[32x16] + C)
  return __builtin_amdgcn_wmma_f32_16x16x32_bf16(false, a, false, b,
                                                 (short)0, c, false, false);
}

__device__ __forceinline__ v8f v8f_zero() {
  v8f z = {0.f, 0.f, 0.f, 0.f, 0.f, 0.f, 0.f, 0.f};
  return z;
}

// xor-butterfly within each 16-lane group (wave32): VALU permlane if available
__device__ __forceinline__ float lane16_xor(float x, int st) {
  const unsigned sel0[4] = {0x67452301u, 0x54761032u, 0x32107654u, 0xFEDCBA98u};
  const unsigned sel1[4] = {0xEFCDAB89u, 0xDCFE98BAu, 0xBA98FEDCu, 0x76543210u};
#if __has_builtin(__builtin_amdgcn_permlane16)
  unsigned u = __builtin_bit_cast(unsigned, x);
  unsigned r = __builtin_amdgcn_permlane16(u, u, sel0[st], sel1[st], false, false);
  return __builtin_bit_cast(float, r);
#else
  (void)sel0; (void)sel1;
  return __shfl_xor(x, 1 << st, 32);
#endif
}

union V16U { v16bf v; v8bf h[2]; };

// ---------------------------------------------------------------------------
// Generic 4096x1024x1024 GEMM: Out = (X * W + bias) * oscale
//   IN_BF16: X is bf16 (attention output) vs fp32 (raw inputs)
//   MODE 0: write bf16 [B,H,S,HD]   (Q with softmax scale folded in, K)
//   MODE 1: write bf16 [B,H,HD,S]   (V transposed: PV B-fragments contiguous)
//   MODE 2: write fp32 row-major [B*S, E]  (final output projection)
// Block: 256 threads = 8 waves; block tile 128(M) x 64(N); per-wave 32x32
// ---------------------------------------------------------------------------
template <bool IN_BF16, int MODE>
__global__ __launch_bounds__(256)
void gemm_kernel(const void* __restrict__ Xv, const float* __restrict__ W,
                 const float* __restrict__ bias, void* __restrict__ Outv,
                 float oscale)
{
  __shared__ __align__(16) __bf16 As[128][40];  // +8 pad keeps 16B row align
  __shared__ __align__(32) __bf16 Bt[64][32];   // W staged TRANSPOSED

  const int tid  = threadIdx.x;
  const int lane = tid & 31;
  const int wave = tid >> 5;
  const int hlf  = lane >> 4;      // 16-lane group (wave32)
  const int l16  = lane & 15;
  const int wm   = wave >> 1;      // 0..3 : 32-row sub-tile
  const int wn   = wave & 1;       // 0..1 : 32-col sub-tile
  const int Mbase = blockIdx.y * 128;
  const int Nbase = blockIdx.x * 64;

  const float*  Xf = (const float*)Xv;
  const __bf16* Xb = (const __bf16*)Xv;

  v8f c[2][2];
  c[0][0] = v8f_zero(); c[0][1] = v8f_zero();
  c[1][0] = v8f_zero(); c[1][1] = v8f_zero();

  for (int kb = 0; kb < E_; kb += 32) {
    // ---- stage A tile (128 x 32) with 16B vector loads --------------------
    if (IN_BF16) {
      #pragma unroll
      for (int i = tid; i < 128 * 4; i += 256) {        // 512 x (8 bf16)
        int m = i >> 2, k = (i & 3) * 8;
        *(v8bf*)&As[m][k] =
            *(const v8bf*)&Xb[(size_t)(Mbase + m) * E_ + (size_t)(kb + k)];
      }
    } else {
      #pragma unroll
      for (int i = tid; i < 128 * 8; i += 256) {        // 1024 x float4
        int m = i >> 3, k = (i & 7) * 4;
        const float4 xv =
            *(const float4*)&Xf[(size_t)(Mbase + m) * E_ + (size_t)(kb + k)];
        v4bf o;
        o[0] = f2bf(xv.x); o[1] = f2bf(xv.y);
        o[2] = f2bf(xv.z); o[3] = f2bf(xv.w);
        *(v4bf*)&As[m][k] = o;
      }
    }
    // ---- stage B tile transposed: Bt[n][k] = W[kb+k][Nbase+n] -------------
    #pragma unroll
    for (int i = tid; i < 512; i += 256) {              // 512 x float4
      int k = i >> 4, n4 = (i & 15) * 4;
      const float4 wv =
          *(const float4*)&W[(size_t)(kb + k) * E_ + (size_t)(Nbase + n4)];
      Bt[n4 + 0][k] = f2bf(wv.x);
      Bt[n4 + 1][k] = f2bf(wv.y);
      Bt[n4 + 2][k] = f2bf(wv.z);
      Bt[n4 + 3][k] = f2bf(wv.w);
    }
    // prefetch next A tile (gfx1250 global_prefetch_b8 path)
    if (kb + 32 < E_) {
      size_t gi = (size_t)(Mbase + wave * 16 + l16) * E_ + (size_t)(kb + 32);
      __builtin_prefetch(IN_BF16 ? (const void*)&Xb[gi] : (const void*)&Xf[gi], 0, 1);
    }
    __syncthreads();

    // ---- per-wave fragments ----------------------------------------------
    // A frag (16x32 bf16): lanes 0-15 -> K 0..15; 16-31 -> K 16..31
    V16U a0, a1;
    const int ar  = wm * 32 + l16;
    const int k0a = hlf * 16;
    a0.h[0] = *(const v8bf*)&As[ar][k0a];
    a0.h[1] = *(const v8bf*)&As[ar][k0a + 8];
    a1.h[0] = *(const v8bf*)&As[ar + 16][k0a];
    a1.h[1] = *(const v8bf*)&As[ar + 16][k0a + 8];
    // B frag (32x16 bf16): lane = column n; lanes 0-15 K=0..15, 16-31 K=16..31
    v16bf b0 = *(const v16bf*)&Bt[wn * 32 + l16][hlf * 16];
    v16bf b1 = *(const v16bf*)&Bt[wn * 32 + 16 + l16][hlf * 16];

    c[0][0] = wmma_bf16(a0.v, b0, c[0][0]);
    c[0][1] = wmma_bf16(a0.v, b1, c[0][1]);
    c[1][0] = wmma_bf16(a1.v, b0, c[1][0]);
    c[1][1] = wmma_bf16(a1.v, b1, c[1][1]);
    __syncthreads();
  }

  // ---- epilogue: C layout (n = l16, m = r + 8*hlf) ------------------------
  #pragma unroll
  for (int mi = 0; mi < 2; ++mi) {
    #pragma unroll
    for (int nj = 0; nj < 2; ++nj) {
      const int coln = Nbase + wn * 32 + nj * 16 + l16;
      const float bv = bias[coln];
      #pragma unroll
      for (int r = 0; r < 8; ++r) {
        const int rowm = Mbase + wm * 32 + mi * 16 + r + hlf * 8;
        const float val = (c[mi][nj][r] + bv) * oscale;
        if (MODE == 2) {
          ((float*)Outv)[(size_t)rowm * E_ + coln] = val;
        } else {
          const int bb = rowm >> 11, s = rowm & (S_ - 1);
          const int hh = coln >> 6,  hd = coln & (HD_ - 1);
          const size_t off = (MODE == 0)
              ? ((size_t)((bb * H_ + hh) * S_ + s)) * HD_ + hd   // [B,H,S,HD]
              : ((size_t)((bb * H_ + hh) * HD_ + hd)) * S_ + s;  // [B,H,HD,S]
          ((__bf16*)Outv)[off] = f2bf(val);
        }
      }
    }
  }
}

// ---------------------------------------------------------------------------
// Flash attention, causal, exp2-domain. Grid: (B*H, S/128). Block 256 = 8
// waves; each wave owns a 16-row query tile and streams 32-key tiles only up
// to its own causal limit. Per tile: 4 wmma (QK^T) + 4 wmma (P*V) + 1 wmma
// (ones column -> row-sum l). Only the single diagonal tile applies the mask.
// ---------------------------------------------------------------------------
__global__ __launch_bounds__(256)
void attn_kernel(const __bf16* __restrict__ Qh, const __bf16* __restrict__ Kh,
                 const __bf16* __restrict__ Vt, __bf16* __restrict__ Attn)
{
  __shared__ __align__(16) __bf16 Ps[8][16][40];  // per-wave P scratch

  const int tid  = threadIdx.x;
  const int lane = tid & 31;
  const int wave = tid >> 5;
  const int hlf  = lane >> 4;
  const int l16  = lane & 15;
  const int bh    = blockIdx.x;            // b*H + h
  const int qbase = blockIdx.y * 128 + wave * 16;

  const __bf16* Qp = Qh + (size_t)bh * S_ * HD_;
  const __bf16* Kp = Kh + (size_t)bh * S_ * HD_;
  const __bf16* Vp = Vt + (size_t)bh * HD_ * S_;   // [HD][S]

  // Q fragments (already scaled by 1/sqrt(HD)*log2e at projection time)
  V16U qa[2];
  #pragma unroll
  for (int kk = 0; kk < 2; ++kk) {
    size_t base = (size_t)(qbase + l16) * HD_ + (size_t)(kk * 32 + hlf * 16);
    qa[kk].h[0] = *(const v8bf*)&Qp[base];
    qa[kk].h[1] = *(const v8bf*)&Qp[base + 8];
  }

  // all-ones B fragment (for the l row-sum WMMA)
  V16U onesu;
  #pragma unroll
  for (int i = 0; i < 16; ++i) onesu.v[i] = f2bf(1.0f);

  float m_run[8];
  v8f acc[4], accl;
  #pragma unroll
  for (int r = 0; r < 8; ++r) m_run[r] = NEG_INF;
  #pragma unroll
  for (int nj = 0; nj < 4; ++nj) acc[nj] = v8f_zero();
  accl = v8f_zero();

  auto tile_step = [&](int kt, bool masked) {
    // ---- scores S = Q * K^T (two 16-key n-tiles, K-dim = HD = 2x32) -------
    v8f sacc0 = v8f_zero(), sacc1 = v8f_zero();
    #pragma unroll
    for (int kk = 0; kk < 2; ++kk) {
      const int key = kt * 32 + l16;       // lane = B column = key row
      v16bf bf0 = *(const v16bf*)&Kp[(size_t)key * HD_ +
                                     (size_t)(kk * 32 + hlf * 16)];
      v16bf bf1 = *(const v16bf*)&Kp[(size_t)(key + 16) * HD_ +
                                     (size_t)(kk * 32 + hlf * 16)];
      sacc0 = wmma_bf16(qa[kk].v, bf0, sacc0);
      sacc1 = wmma_bf16(qa[kk].v, bf1, sacc1);
    }

    // ---- (diagonal tile only) causal mask; then row max -------------------
    float rmax[8], sc[8];
    #pragma unroll
    for (int r = 0; r < 8; ++r) {
      float s0 = sacc0[r], s1 = sacc1[r];
      if (masked) {
        const int m   = qbase + r + hlf * 8;
        const int k0i = kt * 32 + l16;
        if (k0i      > m) s0 = NEG_INF;
        if (k0i + 16 > m) s1 = NEG_INF;
        sacc0[r] = s0; sacc1[r] = s1;
      }
      rmax[r] = fmaxf(s0, s1);
    }
    #pragma unroll
    for (int st = 0; st < 4; ++st) {       // 16-lane butterfly (wave32)
      #pragma unroll
      for (int r = 0; r < 8; ++r)
        rmax[r] = fmaxf(rmax[r], lane16_xor(rmax[r], st));
    }

    // ---- online softmax update; write P (bf16) to per-wave LDS ------------
    #pragma unroll
    for (int r = 0; r < 8; ++r) {
      const float mn = fmaxf(m_run[r], rmax[r]);
      sc[r] = fast_exp2(m_run[r] - mn);
      m_run[r] = mn;
      const float p0 = fast_exp2(sacc0[r] - mn);
      const float p1 = fast_exp2(sacc1[r] - mn);
      Ps[wave][r + hlf * 8][l16]      = f2bf(p0);   // C-layout -> row-major
      Ps[wave][r + hlf * 8][16 + l16] = f2bf(p1);
    }
    #pragma unroll
    for (int nj = 0; nj < 4; ++nj) {
      #pragma unroll
      for (int r = 0; r < 8; ++r)
        acc[nj][r] *= sc[r];
    }
    #pragma unroll
    for (int r = 0; r < 8; ++r)
      accl[r] *= sc[r];

    // ---- O += P*V ; l += P*ones (P reloaded from LDS as A-fragment) -------
    V16U pa;
    pa.h[0] = *(const v8bf*)&Ps[wave][l16][hlf * 16];
    pa.h[1] = *(const v8bf*)&Ps[wave][l16][hlf * 16 + 8];
    #pragma unroll
    for (int nj = 0; nj < 4; ++nj) {
      const int hd = nj * 16 + l16;        // lane = B column = hd
      v16bf vb = *(const v16bf*)&Vp[(size_t)hd * S_ +
                                    (size_t)(kt * 32 + hlf * 16)];
      acc[nj] = wmma_bf16(pa.v, vb, acc[nj]);
    }
    accl = wmma_bf16(pa.v, onesu.v, accl);
  };

  const int kt_diag = qbase >> 5;          // tiles strictly below the diagonal
  for (int kt = 0; kt < kt_diag; ++kt) tile_step(kt, false);
  tile_step(kt_diag, true);                // the one tile that needs masking

  // ---- normalize + write merged heads [B,S,E] bf16 ------------------------
  const int bb = bh >> 4, hh = bh & 15;
  #pragma unroll
  for (int nj = 0; nj < 4; ++nj) {
    #pragma unroll
    for (int r = 0; r < 8; ++r) {
      const int m  = qbase + r + hlf * 8;
      const int hd = nj * 16 + l16;
      Attn[((size_t)(bb * S_ + m)) * E_ + (size_t)(hh * HD_ + hd)] =
          f2bf(acc[nj][r] / accl[r]);
    }
  }
}

// ---------------------------------------------------------------------------
extern "C" void kernel_launch(void* const* d_in, const int* in_sizes, int n_in,
                              void* d_out, int out_size, void* d_ws, size_t ws_size,
                              hipStream_t stream)
{
  (void)in_sizes; (void)n_in; (void)out_size; (void)ws_size;
  const float* q  = (const float*)d_in[0];
  const float* k  = (const float*)d_in[1];
  const float* v  = (const float*)d_in[2];
  /* d_in[3] = causal mask: applied analytically in-kernel */
  const float* Wq = (const float*)d_in[4];
  const float* bq = (const float*)d_in[5];
  const float* Wk = (const float*)d_in[6];
  const float* bk = (const float*)d_in[7];
  const float* Wv = (const float*)d_in[8];
  const float* bv = (const float*)d_in[9];
  const float* Wo = (const float*)d_in[10];
  const float* bo = (const float*)d_in[11];

  const size_t NE = (size_t)B_ * S_ * E_;          // 4M elements (8 MB bf16)
  __bf16* Qh = (__bf16*)d_ws;                      // [B,H,S,HD] (pre-scaled)
  __bf16* Kh = Qh + NE;                            // [B,H,S,HD]
  __bf16* Vt = Kh + NE;                            // [B,H,HD,S]
  __bf16* At = Vt + NE;                            // [B,S,E]   (32 MB total)

  // fold 1/sqrt(HD) * log2(e) into Q so attention runs pure exp2
  const float qscale = 0.125f * 1.44269504088896340736f;

  const dim3 gg(E_ / 64, (B_ * S_) / 128);         // (16, 32)
  gemm_kernel<false, 0><<<gg, 256, 0, stream>>>(q, Wq, bq, Qh, qscale);
  gemm_kernel<false, 0><<<gg, 256, 0, stream>>>(k, Wk, bk, Kh, 1.0f);
  gemm_kernel<false, 1><<<gg, 256, 0, stream>>>(v, Wv, bv, Vt, 1.0f);
  attn_kernel<<<dim3(B_ * H_, S_ / 128), 256, 0, stream>>>(Qh, Kh, Vt, At);
  gemm_kernel<true, 2><<<gg, 256, 0, stream>>>(At, Wo, bo, (float*)d_out, 1.0f);
}
